// Seq2SeqEncoder_24438363914967
// MI455X (gfx1250) — compile-verified, matrix operand order checked by reading, not run on hardware
//
#include <hip/hip_runtime.h>

// ---------------------------------------------------------------------------
// Bidirectional 2-layer LSTM encoder for MI455X (gfx1250, wave32, WMMA).
// B=64, S=512, V=32000, E=256, H=512. Gates fused as z = h@Wh + x@Uh + b,
// gate order f,i,o,g (columns [0,512),[512,1024),[1024,1536),[1536,2048)).
// Weights are pre-swizzled into WMMA B-fragment order -> b128 loads.
// ---------------------------------------------------------------------------

#define BB   64
#define SS   512
#define EE   256
#define HH   512
#define G4   2048          // 4*H
#define SB   (SS * BB)     // 32768 rows for the big GEMMs

typedef __attribute__((ext_vector_type(16))) __bf16        v16bf;
typedef __attribute__((ext_vector_type(8)))  float         v8f;
typedef __attribute__((ext_vector_type(8)))  unsigned int  v8u;

union FragU { v16bf bf; v8u u; };

// ---------------- WMMA fragment loaders (doc layouts, wave32) --------------
// A: 16x32 bf16 tile, lane m = lane&15, half = lane>>4.
// vgpr r holds k-pair starting at k = (r>>2)*16 + half*8 + (r&3)*2  -> one
// 32-bit load per vgpr (pairs are contiguous in row-major A).
__device__ __forceinline__ v16bf load_frag_A(const __bf16* __restrict__ p,
                                             int ld, int m, int half) {
  FragU f;
#pragma unroll
  for (int r = 0; r < 8; ++r) {
    int k = ((r >> 2) << 4) | (half << 3) | ((r & 3) << 1);
    f.u[r] = *(const unsigned int*)(p + m * ld + k);
  }
  return f.bf;
}

// B fragments come pre-swizzled: tile-major, then lane-major, 16 contiguous
// bf16 per lane (32 bytes) -> two global_load_b128.
__device__ __forceinline__ v16bf load_frag_Bp(const __bf16* __restrict__ pk,
                                              size_t tile, int lane) {
  return *(const v16bf*)(pk + tile * 512 + lane * 16);
}
// C/D f32: element r -> row m = r + 8*half, col n = lane&15.

// ---------------- weight pre-swizzle ---------------------------------------
// in: f32 row-major [K,N]; out: bf16, out[((kt*ntn+nt)*32+lane)*16+i] =
// B[kt*32 + kof(r,half,i&1)][nt*16 + (lane&15)], r=i>>1.
__global__ void pack_B_kernel(const float* __restrict__ in,
                              __bf16* __restrict__ out, int K, int N) {
  int idx = blockIdx.x * blockDim.x + threadIdx.x;
  if (idx >= K * N) return;
  int i    = idx & 15;
  int lane = (idx >> 4) & 31;
  int tile = idx >> 9;
  int ntn  = N >> 4;
  int kt = tile / ntn, nt = tile % ntn;
  int n = lane & 15, half = lane >> 4;
  int r = i >> 1;
  int k = (kt << 5) + (((r >> 2) << 4) | (half << 3) | ((r & 3) << 1) | (i & 1));
  out[idx] = (__bf16)in[(size_t)k * N + nt * 16 + n];
}

// xs[t][b][e] = embed[src[b][t]][e], bf16
__global__ void gather_kernel(const int* __restrict__ src,
                              const float* __restrict__ embed,
                              __bf16* __restrict__ xs) {
  int i = blockIdx.x * blockDim.x + threadIdx.x;
  if (i >= SS * BB * EE) return;
  int e = i % EE;
  int b = (i / EE) % BB;
  int t = i / (EE * BB);
  int tok = src[b * SS + t];
  xs[i] = (__bf16)embed[(size_t)tok * EE + e];
}

// ---------------- generic bf16 GEMM: C = A[MxK] * Bpk[KxN] (+bias) ---------
// Each wave computes one 16-row M tile x two adjacent 16-col N tiles (two
// independent accumulator chains -> no WMMA->WMMA hazard NOPs, A reused).
// 8 waves per block. K multiple of 32, N multiple of 32. B pre-swizzled.
__global__ __launch_bounds__(256) void gemm_bf16_kernel(
    const __bf16* __restrict__ A, int lda,
    const __bf16* __restrict__ Bpk,
    const float* __restrict__ bias,
    float* __restrict__ Cf, __bf16* __restrict__ Cb, int ldc,
    int M, int N, int K) {
  int lane = threadIdx.x & 31;
  int wave = threadIdx.x >> 5;
  int ntn  = N >> 4;        // 16-wide N tiles
  int npr  = N >> 5;        // N tile pairs per M tile
  int unit = blockIdx.x * 8 + wave;
  if (unit >= (M >> 4) * npr) return;
  int mt  = unit / npr;
  int nt0 = (unit % npr) * 2;          // first of the two N tiles
  int m = lane & 15, half = lane >> 4;

  v8f acc0, acc1;
  float bv0 = bias ? bias[nt0 * 16 + m] : 0.0f;       // col n = lane&15
  float bv1 = bias ? bias[(nt0 + 1) * 16 + m] : 0.0f;
#pragma unroll
  for (int r = 0; r < 8; ++r) { acc0[r] = bv0; acc1[r] = bv1; }

  for (int kt = 0; kt < (K >> 5); ++kt) {
    v16bf a  = load_frag_A(A + (size_t)(mt * 16) * lda + kt * 32, lda, m, half);
    v16bf b0 = load_frag_Bp(Bpk, (size_t)kt * ntn + nt0, lane);
    v16bf b1 = load_frag_Bp(Bpk, (size_t)kt * ntn + nt0 + 1, lane);
    acc0 = __builtin_amdgcn_wmma_f32_16x16x32_bf16(false, a, false, b0,
                                                   (short)0, acc0, false, false);
    acc1 = __builtin_amdgcn_wmma_f32_16x16x32_bf16(false, a, false, b1,
                                                   (short)0, acc1, false, false);
  }
  if (Cf) {
#pragma unroll
    for (int r = 0; r < 8; ++r) {
      size_t row = (size_t)mt * 16 + r + (half << 3);
      Cf[row * ldc + nt0 * 16 + m]        = acc0[r];
      Cf[row * ldc + (nt0 + 1) * 16 + m]  = acc1[r];
    }
  }
  if (Cb) {
#pragma unroll
    for (int r = 0; r < 8; ++r) {
      size_t row = (size_t)mt * 16 + r + (half << 3);
      Cb[row * ldc + nt0 * 16 + m]       = (__bf16)acc0[r];
      Cb[row * ldc + (nt0 + 1) * 16 + m] = (__bf16)acc1[r];
    }
  }
}

// ---------------- recurrent LSTM scan --------------------------------------
// One workgroup per direction (blockIdx.x: 0=fwd, 1=bwd), 1024 threads =
// 32 waves on one WGP.  LDS: h double-buffered bf16 [2][64][512] (128KB)
// + c f32 [64][512] (128KB) = 256KB of the WGP's 320KB.  Wh pre-swizzled.
__global__ __launch_bounds__(1024) void lstm_scan_kernel(
    const __bf16* __restrict__ Wh_f, const float* __restrict__ xU_f,
    const float* __restrict__ h0_f,  const float* __restrict__ c0_f,
    const __bf16* __restrict__ Wh_b, const float* __restrict__ xU_b,
    const float* __restrict__ h0_b,  const float* __restrict__ c0_b,
    float* __restrict__ out_f32,     // [S][B][2H], writes cols dir*H..+H
    __bf16* __restrict__ out_bf16) { // same layout in bf16
  const int dir = blockIdx.x;
  const __bf16* Wh = dir ? Wh_b : Wh_f;   // packed, N=2048 -> ntn=128
  const float*  xU = dir ? xU_b : xU_f;
  const float*  h0 = dir ? h0_b : h0_f;
  const float*  c0 = dir ? c0_b : c0_f;
  const int coloff = dir * HH;
  const bool rev   = (dir == 1);

  extern __shared__ char smem[];
  __bf16* hbuf = (__bf16*)smem;                       // [2][B*H]
  float*  cbuf = (float*)(smem + 2 * BB * HH * 2);    // [B*H]

  for (int i = threadIdx.x; i < BB * HH; i += 1024) {
    hbuf[i] = (__bf16)h0[i];
    cbuf[i] = c0[i];
  }
  __syncthreads();

  const int lane = threadIdx.x & 31;
  const int w    = threadIdx.x >> 5;
  const int m    = lane & 15;
  const int half = lane >> 4;

  for (int t = 0; t < SS; ++t) {
    const int tt = rev ? (SS - 1 - t) : t;
    const __bf16* hcur = hbuf + (t & 1) * (BB * HH);
    __bf16*       hnxt = hbuf + ((t + 1) & 1) * (BB * HH);
    const float*  xu_t = xU + (size_t)tt * BB * G4;

#pragma unroll
    for (int q = 0; q < 4; ++q) {
      const int unit = w * 4 + q;
      const int rt = unit >> 5;     // batch row tile 0..3
      const int jt = unit & 31;     // column tile within a gate, 0..31

      v8f acc[4];
#pragma unroll
      for (int gi = 0; gi < 4; ++gi) {
        const int ncol = gi * HH + jt * 16 + m;
#pragma unroll
        for (int r = 0; r < 8; ++r)
          acc[gi][r] = xu_t[(size_t)(rt * 16 + r + (half << 3)) * G4 + ncol];
      }

      for (int kt = 0; kt < HH / 32; ++kt) {
        v16bf a = load_frag_A(hcur + (rt * 16) * HH + kt * 32, HH, m, half);
#pragma unroll
        for (int gi = 0; gi < 4; ++gi) {
          v16bf b = load_frag_Bp(Wh, (size_t)kt * 128 + gi * 32 + jt, lane);
          acc[gi] = __builtin_amdgcn_wmma_f32_16x16x32_bf16(
              false, a, false, b, (short)0, acc[gi], false, false);
        }
      }

      // gates + cell/state update (elementwise across the 4 fragments)
      float hnv[8];
#pragma unroll
      for (int r = 0; r < 8; ++r) {
        const int row = rt * 16 + r + (half << 3);
        const int col = jt * 16 + m;
        const float fg = 1.0f / (1.0f + __expf(-acc[0][r]));
        const float ig = 1.0f / (1.0f + __expf(-acc[1][r]));
        const float og = 1.0f / (1.0f + __expf(-acc[2][r]));
        const float gg = tanhf(acc[3][r]);
        const float cn = fg * cbuf[row * HH + col] + ig * gg;
        const float hn = og * tanhf(cn);
        cbuf[row * HH + col] = cn;
        hnxt[row * HH + col] = (__bf16)hn;
        hnv[r] = hn;
      }
      const size_t obase = (size_t)tt * BB * (2 * HH) + coloff + jt * 16 + m;
      if (out_f32) {
#pragma unroll
        for (int r = 0; r < 8; ++r)
          out_f32[obase + (size_t)(rt * 16 + r + (half << 3)) * (2 * HH)] =
              hnv[r];
      }
      if (out_bf16) {
#pragma unroll
        for (int r = 0; r < 8; ++r)
          out_bf16[obase + (size_t)(rt * 16 + r + (half << 3)) * (2 * HH)] =
              (__bf16)hnv[r];
      }
    }
    __syncthreads();
  }
}

// ---------------------------------------------------------------------------
extern "C" void kernel_launch(void* const* d_in, const int* in_sizes, int n_in,
                              void* d_out, int out_size, void* d_ws, size_t ws_size,
                              hipStream_t stream) {
  (void)in_sizes; (void)n_in; (void)out_size; (void)ws_size;

  const int*   src    = (const int*)d_in[0];
  const float* embed  = (const float*)d_in[1];
  const float* Wh0f   = (const float*)d_in[2];
  const float* Uh0f   = (const float*)d_in[3];
  const float* b0f    = (const float*)d_in[4];
  const float* Wh0b   = (const float*)d_in[5];
  const float* Uh0b   = (const float*)d_in[6];
  const float* b0b    = (const float*)d_in[7];
  const float* Wy0    = (const float*)d_in[8];
  const float* by0    = (const float*)d_in[9];
  const float* Wh1f   = (const float*)d_in[10];
  const float* Uh1f   = (const float*)d_in[11];
  const float* b1f    = (const float*)d_in[12];
  const float* Wh1b   = (const float*)d_in[13];
  const float* Uh1b   = (const float*)d_in[14];
  const float* b1b    = (const float*)d_in[15];
  const float* Wy1    = (const float*)d_in[16];
  const float* by1    = (const float*)d_in[17];
  const float* h0_fwd = (const float*)d_in[18];
  const float* c0_fwd = (const float*)d_in[19];
  const float* h0_bwd = (const float*)d_in[20];
  const float* c0_bwd = (const float*)d_in[21];
  float* out = (float*)d_out;

  // bump allocator in workspace
  char* ws = (char*)d_ws;
  size_t off = 0;
  auto alloc = [&](size_t bytes) -> void* {
    off = (off + 255) & ~(size_t)255;
    void* p = ws + off;
    off += bytes;
    return p;
  };

  __bf16* xs   = (__bf16*)alloc((size_t)SB * 512 * 2); // layer0 (lda=256) then xs1 (lda=512)
  __bf16* hcat = (__bf16*)alloc((size_t)SB * 1024 * 2);
  float*  xUf  = (float*)alloc((size_t)SB * G4 * 4);
  float*  xUb  = (float*)alloc((size_t)SB * G4 * 4);
  __bf16* wh0f = (__bf16*)alloc((size_t)HH * G4 * 2);
  __bf16* wh0b = (__bf16*)alloc((size_t)HH * G4 * 2);
  __bf16* uh0f = (__bf16*)alloc((size_t)EE * G4 * 2);
  __bf16* uh0b = (__bf16*)alloc((size_t)EE * G4 * 2);
  __bf16* wy0b = (__bf16*)alloc((size_t)1024 * 512 * 2);
  __bf16* wh1f = (__bf16*)alloc((size_t)HH * G4 * 2);
  __bf16* wh1b = (__bf16*)alloc((size_t)HH * G4 * 2);
  __bf16* uh1f = (__bf16*)alloc((size_t)HH * G4 * 2);
  __bf16* uh1b = (__bf16*)alloc((size_t)HH * G4 * 2);
  __bf16* wy1b = (__bf16*)alloc((size_t)1024 * 512 * 2);

  auto pack = [&](const float* in, __bf16* o, int K, int N) {
    pack_B_kernel<<<(K * N + 255) / 256, 256, 0, stream>>>(in, o, K, N);
  };
  pack(Wh0f, wh0f, HH, G4);  pack(Wh0b, wh0b, HH, G4);
  pack(Uh0f, uh0f, EE, G4);  pack(Uh0b, uh0b, EE, G4);
  pack(Wy0,  wy0b, 1024, 512);
  pack(Wh1f, wh1f, HH, G4);  pack(Wh1b, wh1b, HH, G4);
  pack(Uh1f, uh1f, HH, G4);  pack(Uh1b, uh1b, HH, G4);
  pack(Wy1,  wy1b, 1024, 512);

  // embedding gather -> xs bf16 [S,B,E]
  gather_kernel<<<(SS * BB * EE + 255) / 256, 256, 0, stream>>>(src, embed, xs);

  auto gemm = [&](const __bf16* A, int lda, const __bf16* Bpk,
                  const float* bias, float* Cf, __bf16* Cb, int ldc,
                  int M, int N, int K) {
    int units = (M >> 4) * (N >> 5);   // one wave per (16M x 32N) unit
    gemm_bf16_kernel<<<(units + 7) / 8, 256, 0, stream>>>(
        A, lda, Bpk, bias, Cf, Cb, ldc, M, N, K);
  };

  const size_t smem = 2 * BB * HH * 2 + BB * HH * 4;  // 256 KB

  // ---- layer 0 ----
  gemm(xs, EE, uh0f, b0f, xUf, nullptr, G4, SB, G4, EE);
  gemm(xs, EE, uh0b, b0b, xUb, nullptr, G4, SB, G4, EE);
  lstm_scan_kernel<<<2, 1024, smem, stream>>>(
      wh0f, xUf, h0_fwd + 0 * BB * HH, c0_fwd + 0 * BB * HH,
      wh0b, xUb, h0_bwd + 0 * BB * HH, c0_bwd + 0 * BB * HH,
      nullptr, hcat);
  // projection: xs1 = hcat @ Wy0 + by0   [SB,1024]x[1024,512] -> bf16
  gemm(hcat, 1024, wy0b, by0, nullptr, xs, 512, SB, 512, 1024);

  // ---- layer 1 ----
  gemm(xs, 512, uh1f, b1f, xUf, nullptr, G4, SB, G4, HH);
  gemm(xs, 512, uh1b, b1b, xUb, nullptr, G4, SB, G4, HH);
  lstm_scan_kernel<<<2, 1024, smem, stream>>>(
      wh1f, xUf, h0_fwd + 1 * BB * HH, c0_fwd + 1 * BB * HH,
      wh1b, xUb, h0_bwd + 1 * BB * HH, c0_bwd + 1 * BB * HH,
      out, nullptr);
}
